// MultiHeadAttentionClassical_65481071399924
// MI455X (gfx1250) — matrix operand order repaired and checked by reading, not compiled, not run on hardware
//
#include <hip/hip_runtime.h>

typedef unsigned short u16;
typedef unsigned int   u32;
typedef unsigned long long u64;
typedef u32   u32x4 __attribute__((ext_vector_type(4)));
typedef float v8f   __attribute__((ext_vector_type(8)));
typedef __bf16 v16bf __attribute__((ext_vector_type(16)));

__device__ __forceinline__ u16 f32_to_bf16_rne(float f) {
    u32 u = __float_as_uint(f);
    u32 r = u + 0x7fffu + ((u >> 16) & 1u);   // round-to-nearest-even
    return (u16)(r >> 16);
}
__device__ __forceinline__ float bf16_to_f32(u16 h) {
    return __uint_as_float(((u32)h) << 16);
}

union FragU { v16bf v; u32x4 q[2]; u16 s[16]; };
union PackU { u32x4 q; u16 s[8]; };

// CDNA5 async global->LDS copy (16B per lane), tracked by ASYNCcnt.
__device__ __forceinline__ void async_copy_b128(u32 lds_off, const u16* gp) {
    asm volatile("global_load_async_to_lds_b128 %0, %1, off"
                 :: "v"(lds_off), "v"((u64)(size_t)gp)
                 : "memory");
}
__device__ __forceinline__ void wait_async0() {
    asm volatile("s_wait_asynccnt 0x0" ::: "memory");
}

// ---------------------------------------------------------------------------
// bf16 WMMA GEMM: C = A[M,K] * B[N,K]^T(rowwise-k) + epilogue.
// B is ALWAYS given K-contiguous ([N,K] row-major) -> b128 async staging only.
//   MODE 0 : f32 out,  += bias[col]                      (final projection)
//   MODE 1 : bf16 out, += bias[col]                      (Q/K projections)
//   MODE 2 : bf16 out, exp(scale*acc)                    (unnormalized probs)
//   MODE 3 : bf16 out, acc / rowdiv[bz*M+row]            (P'V, deferred norm)
//   MODE 4 : bf16 out TRANSPOSED, += bias[col]           (V projection -> V^T)
// Block 128x128, 8 waves (2 M x 4 N), wave tile 64x32, K step 64,
// double-buffered LDS filled by global_load_async_to_lds_b128.
// All dims are multiples of tile sizes -> no bounds checks.
// ---------------------------------------------------------------------------
template <int MODE>
__global__ __launch_bounds__(256) void gemm_bf16_wmma(
    const u16* __restrict__ A, long long aBatch, int lda,
    const u16* __restrict__ B, long long bBatch, int ldb,
    void* __restrict__ Cv, long long cBatch, int ldc,
    const float* __restrict__ bias, const float* __restrict__ rowdiv,
    int M, int N, int K, float scale)
{
    constexpr int LDS_LD = 72;                 // 64 data + 8 pad (144B rows, 16B-aligned)
    __shared__ __align__(16) u16 As[2][128 * LDS_LD];
    __shared__ __align__(16) u16 Bs[2][128 * LDS_LD];

    const int bz = blockIdx.z;
    A += (long long)bz * aBatch;
    B += (long long)bz * bBatch;

    const int tid  = threadIdx.x;
    const int lane = tid & 31;
    const int wave = tid >> 5;
    const int wm   = wave >> 2;                // 0..1  (64 rows each)
    const int wn   = wave & 3;                 // 0..3  (32 cols each)
    const int ln16 = lane & 15;
    const int a_kb = (lane >> 4) * 8;
    const int b_kb = (lane >> 4) * 8;

    const int block_m = blockIdx.y * 128;
    const int block_n = blockIdx.x * 128;

    // staging assignment: thread covers half a 64-wide row: 4 async b128 each
    const int st_row = tid >> 1;               // 0..127
    const int st_kb  = (tid & 1) * 32;         // 0,32

    auto stage_tiles = [&](int k0, int buf) {
        const u16* ga = A + (size_t)(block_m + st_row) * lda + k0 + st_kb;
        const u16* gb = B + (size_t)(block_n + st_row) * ldb + k0 + st_kb;
        u32 la = (u32)(size_t)(&As[buf][st_row * LDS_LD + st_kb]);
        u32 lb = (u32)(size_t)(&Bs[buf][st_row * LDS_LD + st_kb]);
        #pragma unroll
        for (int c = 0; c < 4; ++c) {
            async_copy_b128(la + c * 16, ga + c * 8);
            async_copy_b128(lb + c * 16, gb + c * 8);
        }
    };

    v8f acc[4][2] = {};

    stage_tiles(0, 0);
    for (int k0 = 0; k0 < K; k0 += 64) {
        const int cur = (k0 >> 6) & 1;
        wait_async0();
        __syncthreads();
        if (k0 + 64 < K) stage_tiles(k0 + 64, cur ^ 1);

        #pragma unroll
        for (int kc2 = 0; kc2 < 64; kc2 += 32) {
            FragU af[4], bfg[2];
            #pragma unroll
            for (int i = 0; i < 4; ++i) {
                const u16* p = &As[cur][(wm * 64 + i * 16 + ln16) * LDS_LD + kc2 + a_kb];
                af[i].q[0] = *(const u32x4*)(p);
                af[i].q[1] = *(const u32x4*)(p + 16);
            }
            #pragma unroll
            for (int j = 0; j < 2; ++j) {
                const u16* p = &Bs[cur][(wn * 32 + j * 16 + ln16) * LDS_LD + kc2 + b_kb];
                bfg[j].q[0] = *(const u32x4*)(p);
                bfg[j].q[1] = *(const u32x4*)(p + 16);
            }
            #pragma unroll
            for (int i = 0; i < 4; ++i)
                #pragma unroll
                for (int j = 0; j < 2; ++j)
                    acc[i][j] = __builtin_amdgcn_wmma_f32_16x16x32_bf16(
                        false, af[i].v, false, bfg[j].v, (short)0, acc[i][j], false, false);
        }
        __syncthreads();
    }

    // ---- epilogue ----
    #pragma unroll
    for (int i = 0; i < 4; ++i) {
        #pragma unroll
        for (int j = 0; j < 2; ++j) {
            const int m0  = block_m + wm * 64 + i * 16 + (lane >> 4) * 8;
            const int col = block_n + wn * 32 + j * 16 + ln16;
            if (MODE == 4) {
                // transposed bf16 store: 8 consecutive rows for one col = 16B
                PackU pk;
                const float b = bias[col];
                #pragma unroll
                for (int r = 0; r < 8; ++r)
                    pk.s[r] = f32_to_bf16_rne(acc[i][j][r] + b);
                u16* C = (u16*)Cv;
                *(u32x4*)(&C[(size_t)col * ldc + m0]) = pk.q;
            } else {
                #pragma unroll
                for (int r = 0; r < 8; ++r) {
                    const int row = m0 + r;
                    float v = acc[i][j][r];
                    if (MODE == 0) {
                        float* C = (float*)Cv + (long long)bz * cBatch;
                        C[(size_t)row * ldc + col] = v + bias[col];
                    } else if (MODE == 1) {
                        u16* C = (u16*)Cv + (long long)bz * cBatch;
                        C[(size_t)row * ldc + col] = f32_to_bf16_rne(v + bias[col]);
                    } else if (MODE == 2) {
                        u16* C = (u16*)Cv + (long long)bz * cBatch;
                        C[(size_t)row * ldc + col] = f32_to_bf16_rne(__expf(v * scale));
                    } else {
                        u16* C = (u16*)Cv + (long long)bz * cBatch;
                        const float d = rowdiv[(long long)bz * M + row];
                        C[(size_t)row * ldc + col] = f32_to_bf16_rne(v / d);
                    }
                }
            }
        }
    }
}

// ---------------------------------------------------------------------------
__global__ void cvt_f32_bf16(const float* __restrict__ in, u16* __restrict__ out, int n) {
    int i = blockIdx.x * blockDim.x + threadIdx.x;
    const int stride = gridDim.x * blockDim.x;
    for (; i < n; i += stride) out[i] = f32_to_bf16_rne(in[i]);
}

// W[K,N] f32  ->  Wt[N,K] bf16 (output-coalesced; 2MB per matrix, one-shot)
__global__ void cvt_transpose_bf16(const float* __restrict__ in, u16* __restrict__ out,
                                   int Kd, int Nd) {
    size_t i = (size_t)blockIdx.x * blockDim.x + threadIdx.x;
    const size_t stride = (size_t)gridDim.x * blockDim.x;
    const size_t total = (size_t)Kd * Nd;
    for (; i < total; i += stride) {
        const size_t n = i / Kd, k = i % Kd;
        out[i] = f32_to_bf16_rne(in[k * (size_t)Nd + n]);
    }
}

__global__ __launch_bounds__(256) void rowsum_bf16(
    const u16* __restrict__ P, float* __restrict__ out, int ncols) {
    __shared__ float red[256];
    const size_t row = blockIdx.x;
    const u16* p = P + row * (size_t)ncols;
    float s = 0.f;
    for (int c = threadIdx.x * 8; c < ncols; c += 256 * 8) {
        PackU v;
        v.q = *(const u32x4*)(p + c);
        #pragma unroll
        for (int r = 0; r < 8; ++r) s += bf16_to_f32(v.s[r]);
    }
    red[threadIdx.x] = s;
    __syncthreads();
    for (int off = 128; off > 0; off >>= 1) {
        if ((int)threadIdx.x < off) red[threadIdx.x] += red[threadIdx.x + off];
        __syncthreads();
    }
    if (threadIdx.x == 0) out[row] = red[0];
}

// ---------------------------------------------------------------------------
extern "C" void kernel_launch(void* const* d_in, const int* in_sizes, int n_in,
                              void* d_out, int out_size, void* d_ws, size_t ws_size,
                              hipStream_t stream) {
    const int Bb = 4, S = 2048, D = 1024;
    const int MS = Bb * S;                      // 8192 folded rows
    const float SCALE = 0.125f;                 // 1/sqrt(1024/16)

    const float* x  = (const float*)d_in[0];
    const float* Wq = (const float*)d_in[1];
    const float* bq = (const float*)d_in[2];
    const float* Wk = (const float*)d_in[3];
    const float* bk = (const float*)d_in[4];
    const float* Wv = (const float*)d_in[5];
    const float* bv = (const float*)d_in[6];
    const float* Wo = (const float*)d_in[7];
    const float* bo = (const float*)d_in[8];

    char* ws = (char*)d_ws;
    size_t off = 0;
    auto take = [&](size_t bytes) -> char* {
        char* p = ws + off;
        off += (bytes + 255) & ~(size_t)255;
        return p;
    };

    u16*   xb   = (u16*)take((size_t)MS * D * 2);     // x, bf16 [8192,1024]
    u16*   wqt  = (u16*)take((size_t)D * D * 2);      // Wq^T bf16 [N,K]
    u16*   wkt  = (u16*)take((size_t)D * D * 2);
    u16*   wvt  = (u16*)take((size_t)D * D * 2);
    u16*   wot  = (u16*)take((size_t)D * D * 2);
    u16*   qb   = (u16*)take((size_t)MS * D * 2);     // Q bf16 [B*S, D]
    u16*   kbuf = (u16*)take((size_t)MS * D * 2);     // K bf16 [B*S, D]
    u16*   vtb  = (u16*)take((size_t)MS * D * 2);     // V^T bf16 [D, B*S]
    u16*   Pb   = (u16*)take((size_t)Bb * S * S * 2); // exp-scores bf16
    float* lrow = (float*)take((size_t)Bb * S * 4);   // softmax denominators
    u16*   ob   = (u16*)take((size_t)MS * D * 2);     // attn out bf16

    // 1) conversions (x plain; weights converted + transposed to [N,K])
    cvt_f32_bf16<<<2048, 256, 0, stream>>>(x, xb, MS * D);
    cvt_transpose_bf16<<<512, 256, 0, stream>>>(Wq, wqt, D, D);
    cvt_transpose_bf16<<<512, 256, 0, stream>>>(Wk, wkt, D, D);
    cvt_transpose_bf16<<<512, 256, 0, stream>>>(Wv, wvt, D, D);
    cvt_transpose_bf16<<<512, 256, 0, stream>>>(Wo, wot, D, D);

    // 2) projections: [8192,1024] x [1024,1024]
    dim3 gProj(D / 128, MS / 128, 1);
    gemm_bf16_wmma<1><<<gProj, 256, 0, stream>>>(
        xb, 0, D, wqt, 0, D, qb, 0, D, bq, nullptr, MS, D, D, 0.f);
    gemm_bf16_wmma<1><<<gProj, 256, 0, stream>>>(
        xb, 0, D, wkt, 0, D, kbuf, 0, D, bk, nullptr, MS, D, D, 0.f);
    // V projection written TRANSPOSED: Vt[d][b*S+s], ldc = 8192
    gemm_bf16_wmma<4><<<gProj, 256, 0, stream>>>(
        xb, 0, D, wvt, 0, D, vtb, 0, MS, bv, nullptr, MS, D, D, 0.f);

    // 3) scores: per batch Q.K^T (K rows are already [N,K]); exp(scale*s) -> P'
    dim3 gS(S / 128, S / 128, Bb);
    gemm_bf16_wmma<2><<<gS, 256, 0, stream>>>(
        qb, (long long)S * D, D, kbuf, (long long)S * D, D,
        Pb, (long long)S * S, S, nullptr, nullptr, S, S, D, SCALE);

    // 4) softmax denominators
    rowsum_bf16<<<Bb * S, 256, 0, stream>>>(Pb, lrow, S);

    // 5) out_att = (P' @ V) / l ; B = V^T slice: ldb = 8192, batch offset 2048
    dim3 gAV(D / 128, S / 128, Bb);
    gemm_bf16_wmma<3><<<gAV, 256, 0, stream>>>(
        Pb, (long long)S * S, S, vtb, (long long)S, MS,
        ob, (long long)S * D, D, nullptr, lrow, S, D, S, 0.f);

    // 6) final projection -> f32 d_out
    dim3 gO(D / 128, MS / 128, 1);
    gemm_bf16_wmma<0><<<gO, 256, 0, stream>>>(
        ob, 0, D, wot, 0, D, d_out, 0, D, bo, nullptr, MS, D, D, 0.f);
}
